// MoELayer_83743272338044
// MI455X (gfx1250) — compile-verified, matrix operand order checked by reading, not run on hardware
//
#include <hip/hip_runtime.h>
#include <hip/hip_bf16.h>
#include <stdint.h>

// ---------------------------------------------------------------------------
// MoE (top-2 of 8 experts), D=1024, F=4096, T=8192 tokens, on MI455X gfx1250.
// bf16 WMMA path: v_wmma_f32_16x16x32_bf16. Weights converted once to bf16
// (134MB -> resident in 192MB L2), routed expert GEMMs fused (GEMM1 + SiLU +
// GEMM2) with the 16x4096 hidden tile held in LDS (128KB of the 320KB/WGP).
// LDS tiles use an XOR (chunk ^ row) swizzle so A-fragment ds_load_b128 from
// 16 rows hit all 64 banks (row strides are 0 mod 256B -> 16-way conflicts
// without it), while keeping LDS at 160KB = 2 workgroups per WGP.
// ---------------------------------------------------------------------------

#define T_TOK   8192
#define DMODEL  1024
#define DFF     4096
#define NEXP    8

typedef __attribute__((ext_vector_type(16))) __bf16        v16bf;
typedef __attribute__((ext_vector_type(8)))  float         v8f;
typedef __attribute__((ext_vector_type(4)))  unsigned int  v4u;

union Frag { v4u q[2]; v16bf v; };

__device__ __forceinline__ unsigned short f2bf(float f) {
  unsigned int u = __float_as_uint(f);
  return (unsigned short)((u + 0x7FFFu + ((u >> 16) & 1u)) >> 16); // RNE
}
__device__ __forceinline__ unsigned int pack2(float a, float b) {
  return (unsigned int)f2bf(a) | ((unsigned int)f2bf(b) << 16);
}
// silu(v) = v / (1+e^-v) via v_rcp_f32 (no v_div_scale/div_fmas chains)
__device__ __forceinline__ float silu_f(float v) {
  return v * __builtin_amdgcn_rcpf(1.0f + __expf(-v));
}
__device__ __forceinline__ v8f wmma_bf16(const Frag& a, const Frag& b, v8f c) {
  return __builtin_amdgcn_wmma_f32_16x16x32_bf16(false, a.v, false, b.v,
                                                 (short)0, c, false, false);
}

// A-fragment load from a swizzled LDS tile.
// Element (row, bf16-col c) lives at u32 index:
//   row*rowStrideU32 + (((c>>3) ^ row) << 2) + ((c>>1) & 3)
// Fragment chunks for (k0, g): chunk1 = (k0>>3)+g, chunk2 = chunk1 ^ 2.
__device__ __forceinline__ void load_a_frag(const unsigned int* lds,
                                            int rowStrideU32, int row, int g,
                                            int k0, Frag& a) {
  int c1 = (((k0 >> 3) + g) ^ row) << 2;
  int c2 = c1 ^ 8;                       // (chunk ^ 2) << 2
  a.q[0] = *(const v4u*)&lds[row * rowStrideU32 + c1];
  a.q[1] = *(const v4u*)&lds[row * rowStrideU32 + c2];
}

// ---------------------------------------------------------------------------
// Kernel 1: fp32 -> bf16 convert + transpose.  in: [E][R][C] f32 (row-major),
// out: [E][C][R] bf16, so WMMA B-fragments load contiguously along k.
// ---------------------------------------------------------------------------
__global__ __launch_bounds__(256) void transpose_cvt_kernel(
    const float* __restrict__ in, unsigned short* __restrict__ out,
    int R, int C) {
  __shared__ unsigned short tile[32][33];
  size_t eoff = (size_t)blockIdx.z * (size_t)R * (size_t)C;
  in  += eoff;
  out += eoff;
  int c0 = blockIdx.x * 32, r0 = blockIdx.y * 32;
  int tx = threadIdx.x, ty = threadIdx.y;          // block = (32, 8)
#pragma unroll
  for (int i = 0; i < 32; i += 8)
    tile[ty + i][tx] = f2bf(in[(size_t)(r0 + ty + i) * C + (c0 + tx)]);
  __syncthreads();
#pragma unroll
  for (int i = 0; i < 32; i += 8)
    out[(size_t)(c0 + ty + i) * R + (r0 + tx)] = tile[tx][ty + i];
}

// ---------------------------------------------------------------------------
// Kernel 2: gating.  logits = x @ Wg, top-2, softmax over the 2, scatter
// (token, slot, weight) into per-expert compacted lists via atomic counters.
// Positions are nondeterministic but each token's math is position-invariant,
// so output values are deterministic.  One thread per token; Wg in LDS.
// ---------------------------------------------------------------------------
__global__ __launch_bounds__(256) void gate_kernel(
    const float* __restrict__ x, const float* __restrict__ Wg,
    int* __restrict__ cnt, int* __restrict__ tok,
    int* __restrict__ slot, float* __restrict__ wgt) {
  __shared__ float wg[DMODEL * NEXP];              // 32 KB
  int tid = threadIdx.x;
  for (int i = tid; i < DMODEL * NEXP; i += 256) wg[i] = Wg[i];
  __syncthreads();

  int t = blockIdx.x * 256 + tid;
  float acc[NEXP];
#pragma unroll
  for (int e = 0; e < NEXP; ++e) acc[e] = 0.0f;

  const float4* xp = (const float4*)(x + (size_t)t * DMODEL);
  for (int d4 = 0; d4 < DMODEL / 4; ++d4) {
    float4 v = xp[d4];
    int d = d4 * 4;
#pragma unroll
    for (int e = 0; e < NEXP; ++e) {
      acc[e] += v.x * wg[(d + 0) * NEXP + e] + v.y * wg[(d + 1) * NEXP + e]
              + v.z * wg[(d + 2) * NEXP + e] + v.w * wg[(d + 3) * NEXP + e];
    }
  }
  // top-2 (first occurrence wins ties, matching lax.top_k)
  int e0 = 0; float v0 = acc[0];
#pragma unroll
  for (int e = 1; e < NEXP; ++e) if (acc[e] > v0) { v0 = acc[e]; e0 = e; }
  int e1 = -1; float v1 = -3.4e38f;
#pragma unroll
  for (int e = 0; e < NEXP; ++e)
    if (e != e0 && acc[e] > v1) { v1 = acc[e]; e1 = e; }
  float r  = __expf(v1 - v0);          // <= 1, stable
  float w0 = __builtin_amdgcn_rcpf(1.0f + r);
  float w1 = 1.0f - w0;

  int p0 = atomicAdd(&cnt[e0], 1);
  tok[e0 * T_TOK + p0] = t; slot[e0 * T_TOK + p0] = 0; wgt[e0 * T_TOK + p0] = w0;
  int p1 = atomicAdd(&cnt[e1], 1);
  tok[e1 * T_TOK + p1] = t; slot[e1 * T_TOK + p1] = 1; wgt[e1 * T_TOK + p1] = w1;
}

// ---------------------------------------------------------------------------
// Kernel 3: fused expert.  Block = 256 threads (8 waves), one 16-token tile
// of one expert.  LDS: x-tile bf16 [16][1024] (32KB) + h-tile bf16 [16][4096]
// (128KB) = 160KB -> 2 workgroups per 320KB WGP.
//   GEMM1: h = silu(x @ W1[e])     per wave: 16x512, K=1024 (bf16 WMMA K=32)
//   GEMM2: y = h @ W2[e]           per wave: 16x128, K=4096
//   store: contrib[slot][token] = gate_w * y   (no atomics, no init needed)
// ---------------------------------------------------------------------------
__global__ __launch_bounds__(256) void moe_expert_kernel(
    const float* __restrict__ x,
    const unsigned short* __restrict__ W1t,   // [E][F][D] bf16 (B^T layout)
    const unsigned short* __restrict__ W2t,   // [E][D][F] bf16 (B^T layout)
    const int* __restrict__ cnt, const int* __restrict__ tok,
    const int* __restrict__ slot, const float* __restrict__ wgt,
    float* __restrict__ contrib) {
  __shared__ unsigned int lds_x[16 * (DMODEL / 2)];  // 32 KB, swizzled
  __shared__ unsigned int lds_h[16 * (DFF / 2)];     // 128 KB, swizzled

  // ---- map blockIdx.x -> (expert e, tile) via tile-count prefix sum ----
  int cl[NEXP];
#pragma unroll
  for (int i = 0; i < NEXP; ++i) cl[i] = cnt[i];
  int total = 0, e = -1, tile = 0;
#pragma unroll
  for (int i = 0; i < NEXP; ++i) {
    int ti = (cl[i] + 15) >> 4;
    if (e < 0 && (int)blockIdx.x < total + ti) { e = i; tile = (int)blockIdx.x - total; }
    total += ti;
  }
  if (e < 0) return;                       // beyond last tile (uniform branch)

  const int base = tile * 16;
  const int cE   = cl[e];
  const int tid  = threadIdx.x;

  // uniform (SGPR) per-expert weight bases -> saddr+voffset global loads
  const unsigned short* W1e = W1t + (size_t)e * DFF * DMODEL;
  const unsigned short* W2e = W2t + (size_t)e * DMODEL * DFF;

  // ---- stage x tile (gather 16 routed tokens) into swizzled LDS bf16 ----
  {
    int r  = tid >> 4;                     // row 0..15
    int cs = (tid & 15) * 64;              // 64 floats per thread
    int tr = (base + r < cE) ? tok[e * T_TOK + base + r] : -1;
    const float4* xp = (tr >= 0) ? (const float4*)(x + (size_t)tr * DMODEL + cs)
                                 : (const float4*)nullptr;
#pragma unroll
    for (int j = 0; j < 64; j += 4) {
      float4 v = (tr >= 0) ? xp[j >> 2] : make_float4(0.f, 0.f, 0.f, 0.f);
      int c  = cs + j;                                  // bf16 column
      int ui = r * (DMODEL / 2) + ((((c >> 3) ^ r) << 2) | ((c >> 1) & 3));
      lds_x[ui + 0] = pack2(v.x, v.y);
      lds_x[ui + 1] = pack2(v.z, v.w);
    }
  }
  __syncthreads();

  const int wave = tid >> 5;
  const int lane = tid & 31;
  const int g    = (lane >> 4) & 1;        // ISA 16-bit A/B k-half select
  const int ln   = lane & 15;              // A: row, B: column

  // ---- GEMM1: h[16, wave*512 .. +512) = silu(x @ W1[e]), bf16 to LDS ----
  for (int ntc = 0; ntc < 4; ++ntc) {      // 4 chunks of 8 n-tiles (A reuse x8)
    v8f acc[8] = {};
    const int n0 = wave * 512 + ntc * 128;
    // per-lane 32-bit element offset; (i*16*DMODEL + k0) goes in the 24-bit imm
    const unsigned boff1 = (unsigned)(n0 + ln) * DMODEL + 8u * g;
    for (int k0 = 0; k0 < DMODEL; k0 += 32) {
      Frag a;
      load_a_frag(lds_x, DMODEL / 2, ln, g, k0, a);
#pragma unroll
      for (int i = 0; i < 8; ++i) {
        Frag b;
        const v4u* pb = (const v4u*)(W1e + boff1 + i * 16 * DMODEL + k0);
        b.q[0] = pb[0]; b.q[1] = pb[2];
        acc[i] = wmma_bf16(a, b, acc[i]);
      }
    }
    // SiLU on f32 accumulators, store swizzled bf16 h tile
    unsigned short* hh = (unsigned short*)lds_h;
#pragma unroll
    for (int i = 0; i < 8; ++i) {
#pragma unroll
      for (int r8 = 0; r8 < 8; ++r8) {
        int m = r8 + 8 * g;
        int n = n0 + i * 16 + ln;
        hh[m * DFF + ((((n >> 3) ^ m) << 3) | (n & 7))] = f2bf(silu_f(acc[i][r8]));
      }
    }
  }
  __syncthreads();

  // ---- GEMM2: y[16, wave*128 .. +128) = h @ W2[e] ----
  v8f oacc[8] = {};
  {
    const unsigned boff2 = (unsigned)(wave * 128 + ln) * DFF + 8u * g;
    for (int k0 = 0; k0 < DFF; k0 += 32) {
      Frag a;
      load_a_frag(lds_h, DFF / 2, ln, g, k0, a);
#pragma unroll
      for (int i = 0; i < 8; ++i) {
        Frag b;
        const v4u* pb = (const v4u*)(W2e + boff2 + i * 16 * DFF + k0);
        b.q[0] = pb[0]; b.q[1] = pb[2];
        oacc[i] = wmma_bf16(a, b, oacc[i]);
      }
    }
  }

  // ---- scale by gate weight, scatter to contrib[slot][token][n] ----
#pragma unroll
  for (int r8 = 0; r8 < 8; ++r8) {
    int m   = r8 + 8 * g;
    int idx = base + m;
    if (idx < cE) {
      int   tk = tok [e * T_TOK + idx];
      int   sl = slot[e * T_TOK + idx];
      float w  = wgt [e * T_TOK + idx];
      float* dst = contrib + ((size_t)sl * T_TOK + tk) * DMODEL;
#pragma unroll
      for (int i = 0; i < 8; ++i) {
        int n = wave * 128 + i * 16 + ln;
        dst[n] = w * oacc[i][r8];
      }
    }
  }
}

// ---------------------------------------------------------------------------
// Kernel 4: out = contrib[slot0] + contrib[slot1]  (fixed order, fully writes
// d_out -> deterministic values, no zero-init of d_out required)
// ---------------------------------------------------------------------------
__global__ __launch_bounds__(256) void combine_kernel(
    const float* __restrict__ contrib, float* __restrict__ out) {
  size_t i = ((size_t)blockIdx.x * 256 + threadIdx.x) * 4;
  const float4 a = *(const float4*)(contrib + i);
  const float4 b = *(const float4*)(contrib + (size_t)T_TOK * DMODEL + i);
  *(float4*)(out + i) = make_float4(a.x + b.x, a.y + b.y, a.z + b.z, a.w + b.w);
}

// ---------------------------------------------------------------------------
// Workspace layout (needs ~193 MB):
//   [0,   64MB)  W1t bf16  [E][F][D]
//   [64,  128MB) W2t bf16  [E][D][F]
//   [128, 192MB) contrib f32 [2][T][D]
//   [192MB ...)  cnt[16] | tok[E*T] | slot[E*T] | wgt[E*T]
// ---------------------------------------------------------------------------
extern "C" void kernel_launch(void* const* d_in, const int* in_sizes, int n_in,
                              void* d_out, int out_size, void* d_ws, size_t ws_size,
                              hipStream_t stream) {
  const float* x  = (const float*)d_in[0];
  const float* Wg = (const float*)d_in[1];
  const float* W1 = (const float*)d_in[2];
  const float* W2 = (const float*)d_in[3];
  float* out = (float*)d_out;

  char* ws = (char*)d_ws;
  unsigned short* W1t = (unsigned short*)(ws);
  unsigned short* W2t = (unsigned short*)(ws + ((size_t)64 << 20));
  float* contrib      = (float*)(ws + ((size_t)128 << 20));
  int*   cnt          = (int*)(ws + ((size_t)192 << 20));
  int*   tok          = cnt + 16;
  int*   slot         = tok + NEXP * T_TOK;
  float* wgt          = (float*)(slot + NEXP * T_TOK);

  hipMemsetAsync(cnt, 0, 16 * sizeof(int), stream);

  // one-time-per-call weight bf16 conversion (L2-resident afterwards)
  transpose_cvt_kernel<<<dim3(DFF / 32, DMODEL / 32, NEXP), dim3(32, 8), 0, stream>>>(
      W1, W1t, DMODEL, DFF);
  transpose_cvt_kernel<<<dim3(DMODEL / 32, DFF / 32, NEXP), dim3(32, 8), 0, stream>>>(
      W2, W2t, DFF, DMODEL);

  gate_kernel<<<T_TOK / 256, 256, 0, stream>>>(x, Wg, cnt, tok, slot, wgt);

  // sum_e ceil(cnt[e]/16) <= 2*T/16 + 8 = 1032 tiles; extra blocks early-exit
  moe_expert_kernel<<<1032, 256, 0, stream>>>(x, W1t, W2t, cnt, tok, slot, wgt,
                                              contrib);

  combine_kernel<<<(T_TOK * DMODEL) / (256 * 4), 256, 0, stream>>>(contrib, out);
}
